// MultiLabelVulnGNN_70626442215629
// MI455X (gfx1250) — matrix operand order repaired and checked by reading, not compiled, not run on hardware
//
#include <hip/hip_runtime.h>

// CDNA5 (gfx1250) wave32 WMMA fragment types
typedef __attribute__((ext_vector_type(16))) __bf16 v16bf;
typedef __attribute__((ext_vector_type(8)))  float  v8f;

#define BN_EPS 1e-5f

// ---------------- degree / norm ----------------
__global__ void k_init_deg(float* dis, int n) {
    int i = blockIdx.x * blockDim.x + threadIdx.x;
    if (i < n) dis[i] = 1.0f;                       // self-loop contributes 1
}

__global__ void k_deg_accum(const long long* __restrict__ ei, float* dis, int E_) {
    int i = blockIdx.x * blockDim.x + threadIdx.x;
    if (i < E_) {
        int dst = (int)ei[(long long)E_ + i];
        unsafeAtomicAdd(dis + dst, 1.0f);
    }
}

__global__ void k_deg_to_dis(float* dis, int n) {
    int i = blockIdx.x * blockDim.x + threadIdx.x;
    if (i < n) dis[i] = rsqrtf(dis[i]);             // D^{-1/2}
}

// ---------------- bf16 WMMA GEMM: Out[M,Nc] = A[M,K] @ W[K,Nc] (+bias, relu) ---
// grid.x = M/(16*MT) row tiles; blockDim.x = 32 * (Nc/16); one wave per N-tile,
// MT M-tiles per block so each B fragment is reused MT times.
// KFULL=true -> K == KTILES*32 exactly (no pad guards, fully unrolled).
template<int KTILES, int MT, bool KFULL>
__global__ void k_gemm_wmma(const float* __restrict__ A, const float* __restrict__ W,
                            const float* __restrict__ bias, float* __restrict__ Out,
                            int K, int Nc, int do_relu)
{
    constexpr int KP = KTILES * 32;
    __shared__ float As[16 * MT][KP];
    const int m0  = blockIdx.x * (16 * MT);
    const int tid = threadIdx.x;

    // cooperative stage of the (16*MT) x KP A tile (zero-padded in K if !KFULL)
    for (int i = tid; i < 16 * MT * KP; i += blockDim.x) {
        int r = i / KP, c = i % KP;                 // KP is constexpr -> shifts
        float v;
        if (KFULL) v = A[(long long)(m0 + r) * KP + c];
        else       v = (c < K) ? A[(long long)(m0 + r) * K + c] : 0.0f;
        As[r][c] = v;
    }
    __syncthreads();

    const int lane = tid & 31;
    const int wave = tid >> 5;
    const int n0   = wave * 16;
    const int mr   = lane & 15;                     // A row / B,C column within tile
    const int hi   = lane >> 4;                     // lane half selects K/M offsets

    v8f acc[MT];
#pragma unroll
    for (int mt = 0; mt < MT; ++mt)
        acc[mt] = (v8f){0.f, 0.f, 0.f, 0.f, 0.f, 0.f, 0.f, 0.f};

#pragma unroll
    for (int kt = 0; kt < KTILES; ++kt) {
        const int k0 = kt * 32;
        // B 32x16 bf16 layout: lane holds N-col (n0+lane%16), elements stripe K in A's order
        v16bf b;
#pragma unroll
        for (int e = 0; e < 16; ++e) {
            const int kb = k0 + (e < 8 ? e : e + 8) + (hi ? 8 : 0);
            b[e] = (KFULL || kb < K) ? (__bf16)W[(long long)kb * Nc + n0 + mr] : (__bf16)0.0f;
        }
#pragma unroll
        for (int mt = 0; mt < MT; ++mt) {
            // A 16x32 bf16 layout: lane<16 -> K 0..7,16..23 ; lane>=16 -> 8..15,24..31
            v16bf a;
#pragma unroll
            for (int e = 0; e < 16; ++e) {
                int kk = (e < 8 ? e : e + 8) + (hi ? 8 : 0);
                a[e] = (__bf16)As[mt * 16 + mr][k0 + kk];
            }
            acc[mt] = __builtin_amdgcn_wmma_f32_16x16x32_bf16(
                          false, a, false, b, (short)0, acc[mt], false, false);
        }
    }

    // C/D layout: VGPR r, lanes 0-15 -> M=r, lanes 16-31 -> M=r+8; N = lane&15
    const int col  = n0 + mr;
    const float bv = bias ? bias[col] : 0.0f;
#pragma unroll
    for (int mt = 0; mt < MT; ++mt) {
#pragma unroll
        for (int r = 0; r < 8; ++r) {
            int   row = m0 + mt * 16 + r + (hi ? 8 : 0);
            float v   = acc[mt][r] + bv;
            if (do_relu) v = fmaxf(v, 0.0f);
            Out[(long long)row * Nc + col] = v;
        }
    }
}

// ---------------- GCN aggregation (H == 128 hard-coded) ----------------
// out[n,h] = tmp[n,h] * dis[n]^2 + b[h]   (float4 over H)
__global__ void k_init_out(const float4* __restrict__ tmp, const float* __restrict__ dis,
                           const float4* __restrict__ b4, float4* __restrict__ out, int total4)
{
    int i = blockIdx.x * blockDim.x + threadIdx.x;
    if (i < total4) {
        int node = i >> 5;                           // 32 float4 per 128-wide row
        float  s  = dis[node]; s *= s;
        float4 t  = tmp[i];
        float4 bb = b4[i & 31];
        out[i] = make_float4(t.x * s + bb.x, t.y * s + bb.y,
                             t.z * s + bb.z, t.w * s + bb.w);
    }
}

// one wave per edge: lane handles 4 features -> coalesced 512B gather of tmp[src]
__global__ void k_scatter(const long long* __restrict__ ei, const float* __restrict__ dis,
                          const float* __restrict__ tmp, float* __restrict__ out, int E_)
{
    long long idx = (long long)blockIdx.x * blockDim.x + threadIdx.x;
    if (idx >= (long long)E_ * 32) return;
    int e  = (int)(idx >> 5);
    int c4 = (int)(idx & 31) << 2;
    int src = (int)ei[e];
    int dst = (int)ei[(long long)E_ + e];
    float norm = dis[src] * dis[dst];
    const float4 v = *(const float4*)(tmp + (long long)src * 128 + c4);
    float* o = out + (long long)dst * 128 + c4;
    unsafeAtomicAdd(o + 0, v.x * norm);
    unsafeAtomicAdd(o + 1, v.y * norm);
    unsafeAtomicAdd(o + 2, v.z * norm);
    unsafeAtomicAdd(o + 3, v.w * norm);
}

// eval-mode BatchNorm + ReLU, in place (H == 128, float4)
__global__ void k_bnrelu(float4* __restrict__ h, const float4* __restrict__ g,
                         const float4* __restrict__ be, const float4* __restrict__ m,
                         const float4* __restrict__ v, int total4)
{
    int i = blockIdx.x * blockDim.x + threadIdx.x;
    if (i < total4) {
        int c = i & 31;
        float4 hv = h[i], gv = g[c], bv = be[c], mv = m[c], vv = v[c];
        float4 r;
        r.x = fmaxf((hv.x - mv.x) * rsqrtf(vv.x + BN_EPS) * gv.x + bv.x, 0.0f);
        r.y = fmaxf((hv.y - mv.y) * rsqrtf(vv.y + BN_EPS) * gv.y + bv.y, 0.0f);
        r.z = fmaxf((hv.z - mv.z) * rsqrtf(vv.z + BN_EPS) * gv.z + bv.z, 0.0f);
        r.w = fmaxf((hv.w - mv.w) * rsqrtf(vv.w + BN_EPS) * gv.w + bv.w, 0.0f);
        h[i] = r;
    }
}

// ---------------- pooling ----------------
__global__ void k_zero(float* p, int n) {
    int i = blockIdx.x * blockDim.x + threadIdx.x;
    if (i < n) p[i] = 0.0f;
}

__global__ void k_pool_accum(const float* __restrict__ h, const long long* __restrict__ batch,
                             float* __restrict__ pooled, float* __restrict__ cnt, int total)
{
    int i = blockIdx.x * blockDim.x + threadIdx.x;
    if (i < total) {
        int node = i >> 7, c = i & 127;
        int g = (int)batch[node];
        unsafeAtomicAdd(pooled + (long long)g * 128 + c, h[i]);
        if (c == 0) unsafeAtomicAdd(cnt + g, 1.0f);
    }
}

__global__ void k_pool_div(float* pooled, const float* __restrict__ cnt, int total) {
    int i = blockIdx.x * blockDim.x + threadIdx.x;
    if (i < total) pooled[i] /= fmaxf(cnt[i >> 7], 1.0f);
}

// ---------------- classifier tail: logits[G,C] = hid[G,K] @ Wc2[K,C] + bc2 ---
__global__ void k_mlp2(const float* __restrict__ hid, const float* __restrict__ W,
                       const float* __restrict__ b, float* __restrict__ out,
                       int G_, int K, int C_)
{
    int i = blockIdx.x * blockDim.x + threadIdx.x;
    if (i < G_ * C_) {
        int g = i / C_, c = i % C_;
        float s = b[c];
        for (int k = 0; k < K; ++k) s += hid[g * K + k] * W[k * C_ + c];
        out[i] = s;
    }
}

extern "C" void kernel_launch(void* const* d_in, const int* in_sizes, int n_in,
                              void* d_out, int out_size, void* d_ws, size_t ws_size,
                              hipStream_t stream)
{
    (void)n_in; (void)ws_size;
    const float*     x     = (const float*)d_in[0];
    const long long* ei    = (const long long*)d_in[1];
    const long long* batch = (const long long*)d_in[2];
    const float* Ws[3]  = {(const float*)d_in[3],  (const float*)d_in[9],  (const float*)d_in[15]};
    const float* bs[3]  = {(const float*)d_in[4],  (const float*)d_in[10], (const float*)d_in[16]};
    const float* gs[3]  = {(const float*)d_in[5],  (const float*)d_in[11], (const float*)d_in[17]};
    const float* bes[3] = {(const float*)d_in[6],  (const float*)d_in[12], (const float*)d_in[18]};
    const float* ms[3]  = {(const float*)d_in[7],  (const float*)d_in[13], (const float*)d_in[19]};
    const float* vs[3]  = {(const float*)d_in[8],  (const float*)d_in[14], (const float*)d_in[20]};
    const float* Wc1 = (const float*)d_in[21];
    const float* bc1 = (const float*)d_in[22];
    const float* Wc2 = (const float*)d_in[23];
    const float* bc2 = (const float*)d_in[24];

    const int N_ = in_sizes[2];          // 100000 (multiple of 32)
    const int E_ = in_sizes[1] / 2;      // 1600000
    const int H_ = in_sizes[4];          // 128
    const int F_ = in_sizes[0] / N_;     // 13
    const int H2 = in_sizes[22];         // 64
    const int C_ = in_sizes[24];         // 10
    const int G_ = out_size / C_;        // 512

    // workspace layout (f32)
    float* buf0   = (float*)d_ws;                       // tmp = h @ W      [N,H]
    float* buf1   = buf0 + (size_t)N_ * H_;             // aggregated/h     [N,H]
    float* dis    = buf1 + (size_t)N_ * H_;             // D^{-1/2}         [N]
    float* pooled = dis + N_;                           // [G,H]
    float* cnt    = pooled + (size_t)G_ * H_;           // [G] (contiguous after pooled)
    float* hid    = cnt + G_;                           // [G,H2]

    const int TB  = 256;
    const int NH  = N_ * H_;
    const int NH4 = NH / 4;

    // degree -> D^{-1/2}
    k_init_deg  <<<(N_ + TB - 1) / TB, TB, 0, stream>>>(dis, N_);
    k_deg_accum <<<(E_ + TB - 1) / TB, TB, 0, stream>>>(ei, dis, E_);
    k_deg_to_dis<<<(N_ + TB - 1) / TB, TB, 0, stream>>>(dis, N_);

    // three GCN layers
    for (int l = 0; l < 3; ++l) {
        if (l == 0) {
            // layer 1: K = 13 padded to 32 (guarded instantiation)
            k_gemm_wmma<1, 2, false><<<N_ / 32, 32 * (H_ / 16), 0, stream>>>(
                x, Ws[0], nullptr, buf0, F_, H_, 0);
        } else {
            // K = 128 exactly: fully unrolled, guard-free
            k_gemm_wmma<4, 2, true><<<N_ / 32, 32 * (H_ / 16), 0, stream>>>(
                buf1, Ws[l], nullptr, buf0, H_, H_, 0);
        }
        k_init_out<<<(NH4 + TB - 1) / TB, TB, 0, stream>>>(
            (const float4*)buf0, dis, (const float4*)bs[l], (float4*)buf1, NH4);
        long long sthreads = (long long)E_ * 32;
        k_scatter<<<(unsigned)((sthreads + TB - 1) / TB), TB, 0, stream>>>(ei, dis, buf0, buf1, E_);
        k_bnrelu<<<(NH4 + TB - 1) / TB, TB, 0, stream>>>(
            (float4*)buf1, (const float4*)gs[l], (const float4*)bes[l],
            (const float4*)ms[l], (const float4*)vs[l], NH4);
    }

    // global mean pool
    k_zero      <<<(G_ * H_ + G_ + TB - 1) / TB, TB, 0, stream>>>(pooled, G_ * H_ + G_);
    k_pool_accum<<<(NH + TB - 1) / TB, TB, 0, stream>>>(buf1, batch, pooled, cnt, NH);
    k_pool_div  <<<(G_ * H_ + TB - 1) / TB, TB, 0, stream>>>(pooled, cnt, G_ * H_);

    // classifier: WMMA for [G,H]@[H,H2] (+bias+relu), then tiny [G,H2]@[H2,C]
    k_gemm_wmma<4, 2, true><<<G_ / 32, 32 * (H2 / 16), 0, stream>>>(
        pooled, Wc1, bc1, hid, H_, H2, 1);
    k_mlp2<<<(G_ * C_ + TB - 1) / TB, TB, 0, stream>>>(hid, Wc2, bc2, (float*)d_out, G_, H2, C_);
}
